// CMLANet_13365938226107
// MI455X (gfx1250) — compile-verified, matrix operand order checked by reading, not run on hardware
//
#include <hip/hip_runtime.h>
#include <hip/hip_bf16.h>

#define BS     4
#define NWORD  2048
#define CS     5
#define DE     300
#define NH     512
#define NT     20
#define NC     5
#define NV     40
#define NIN    1500
#define VOCAB  50000
#define KP     1536          // K padded (1500 -> 1536) for main GEMM
#define G3     (3*NH)        // 1536
#define GV3    (3*NV)        // 120
#define ROWS   (BS*NWORD)    // 8192

typedef __attribute__((ext_vector_type(16))) __bf16 v16bf;
typedef __attribute__((ext_vector_type(8)))  __bf16 v8bf;
typedef __attribute__((ext_vector_type(8)))  float  v8f;

#if defined(__HIP_DEVICE_COMPILE__) && \
    __has_builtin(__builtin_amdgcn_tensor_load_to_lds) && \
    __has_builtin(__builtin_amdgcn_s_wait_tensorcnt)
#define USE_TDM 1
#else
#define USE_TDM 0
#endif

// ---------------------------------------------------------------------------
// WMMA helpers (CDNA5 16x16x32 bf16, f32 accumulate)
// A 16x32 (MxK) fragment, lane L: row = L&15, half = L>>4.
// v16bf elements 0..7  -> K = kb + 8*half + 0..7
// v16bf elements 8..15 -> K = kb + 16 + 8*half + 0..7
// B consumed as NxK row-major with the same loader (row = output col).
// ---------------------------------------------------------------------------
__device__ __forceinline__ v16bf load_frag16x32(const __bf16* base, int ldk,
                                                int row, int hf, int kb) {
  union { v16bf v; v8bf h[2]; } f;
  const __bf16* p = base + (size_t)row * ldk + kb + 8 * hf;
  f.h[0] = *(const v8bf*)(p);
  f.h[1] = *(const v8bf*)(p + 16);
  return f.v;
}

__device__ __forceinline__ v8f wmma_bf16(v16bf a, v16bf b, v8f c) {
  return __builtin_amdgcn_wmma_f32_16x16x32_bf16(false, a, false, b,
                                                 (short)0, c, false, false);
}

__device__ __forceinline__ float sigmoidf_(float x) {
  return 1.0f / (1.0f + __expf(-x));
}

#if USE_TDM
typedef __attribute__((ext_vector_type(4))) unsigned int u32x4;
typedef __attribute__((ext_vector_type(8))) int          i32x8;
typedef __attribute__((ext_vector_type(4))) int          i32x4;

// LDS byte offset of a generic pointer to a __shared__ object.
__device__ __forceinline__ unsigned lds_offset(const void* p) {
  return (unsigned)(size_t)(__attribute__((address_space(3))) const void*)p;
}

// Issue one 2D TDM tile load: rows x 32 bf16 elements, row stride KP elements,
// packed contiguously into LDS at ldsOff.  D# packing per ISA 8.3/8.4:
//  g0: [1:0]=count=1, [63:32]=lds_addr, [120:64]=global_addr, [127:126]=type 2
//  g1: [17:16]=data_size(2B), [79:48]=tensor_dim0, [111:80]=tensor_dim1,
//      [127:112]=tile_dim0, [143:128]=tile_dim1, [207:160]=dim0_stride
// This toolchain exposes the 6-arg builtin:
//   (uint32x4 g0, int32x8 g1, int32x4, int32x4, int32x8, i32 cpol)
// groups beyond g1 are unused for a 2D tile -> zero-filled.
__device__ __forceinline__ void tdm_load_tile(unsigned ldsOff,
                                              unsigned long long gaddr,
                                              int rows) {
  u32x4 g0 = { 1u, ldsOff,
               (unsigned)(gaddr & 0xffffffffull),
               (unsigned)((gaddr >> 32) & 0x01ffffffull) | 0x80000000u };
  i32x8 g1 = { (int)(1u << 16),                         // data_size = 2 bytes
               (int)((unsigned)(KP & 0xffff) << 16),    // tensor_dim0 lo16
               (int)(((unsigned)rows & 0xffffu) << 16), // dim0 hi16=0 | dim1 lo16
               (int)(32u << 16),                        // dim1 hi16=0 | tile_dim0
               rows,                                    // tile_dim1 (tile_dim2=0)
               (int)KP,                                 // dim0_stride lo32
               0, 0 };
  i32x4 z4 = { 0, 0, 0, 0 };
  i32x8 z8 = { 0, 0, 0, 0, 0, 0, 0, 0 };
  __builtin_amdgcn_tensor_load_to_lds(g0, g1, z4, z4, z8, 0);
}
#endif

// ---------------------------------------------------------------------------
// Convert (and K-pad) an f32 matrix to bf16, row-major rows x kout.
// ---------------------------------------------------------------------------
__global__ void cvt_pad_bf16(const float* __restrict__ in, __bf16* __restrict__ out,
                             int rows, int kin, int kout) {
  int idx = blockIdx.x * blockDim.x + threadIdx.x;
  if (idx >= rows * kout) return;
  int r = idx / kout, c = idx - r * kout;
  float v = (c < kin) ? in[(size_t)r * kin + c] : 0.0f;
  out[idx] = (__bf16)v;
}

// ---------------------------------------------------------------------------
// Gather embedding context windows -> x_bf16 (ROWS x KP), zero K-padded.
// ---------------------------------------------------------------------------
__global__ void build_x(const int* __restrict__ ctxw, const int* __restrict__ iemb,
                        const float* __restrict__ emb, const float* __restrict__ pad,
                        const float* __restrict__ pkt, __bf16* __restrict__ x) {
  int rrow = blockIdx.x;                // 0..ROWS-1 = b*NWORD + w
  int b = rrow / NWORD;
  const int* cw = ctxw + (size_t)rrow * CS;
  __bf16* xr = x + (size_t)rrow * KP;
  for (int col = threadIdx.x; col < KP; col += blockDim.x) {
    float v = 0.0f;
    if (col < NIN) {
      int c = col / DE, d = col - c * DE;
      int node = cw[c];
      if (node < NWORD)       v = emb[(size_t)d * VOCAB + iemb[b * NWORD + node]];
      else if (node == NWORD) v = pad[d];
      else                    v = pkt[d];
    }
    xr[col] = (__bf16)v;
  }
}

// ---------------------------------------------------------------------------
// xw = x @ Wih^T + bih.  A: ROWSxKP bf16, B: G3xKP bf16 (NxK), D: ROWSxG3 f32.
// Block = 256 thr (8 waves), block tile 32(M) x 256(N), K-step 32.
// TDM double-buffers the 32x32 A tile and 256x32 B tile into LDS; waves
// consume fragments via ds_load_b128, so WMMA never waits on global latency.
// ---------------------------------------------------------------------------
__global__ void __launch_bounds__(256)
gemm_xw(const __bf16* __restrict__ A, const __bf16* __restrict__ B,
        const float* __restrict__ bias, float* __restrict__ D) {
  __shared__ __bf16 sA[2][32 * 32];
  __shared__ __bf16 sB[2][256 * 32];
  int tid = threadIdx.x;
  int wave = tid >> 5, lane = tid & 31;
  int row = lane & 15, hf = lane >> 4;
  int wm = wave & 1, wn = wave >> 1;
  int mBase = blockIdx.x * 32;
  int nBase = blockIdx.y * 256;

#if USE_TDM
  unsigned offA[2] = { lds_offset(&sA[0][0]), lds_offset(&sA[1][0]) };
  unsigned offB[2] = { lds_offset(&sB[0][0]), lds_offset(&sB[1][0]) };
  unsigned long long gA = (unsigned long long)(size_t)(A + (size_t)mBase * KP);
  unsigned long long gB = (unsigned long long)(size_t)(B + (size_t)nBase * KP);
  if (wave == 0) {
    tdm_load_tile(offA[0], gA, 32);
    tdm_load_tile(offB[0], gB, 256);
    __builtin_amdgcn_s_wait_tensorcnt(0);
  }
  __syncthreads();
#endif

  v8f acc0 = {}, acc1 = {}, acc2 = {}, acc3 = {};
  int ib = 0;
  for (int kb = 0; kb < KP; kb += 32, ib ^= 1) {
#if USE_TDM
    if (wave == 0 && kb + 32 < KP) {      // prefetch next K-step into other buffer
      tdm_load_tile(offA[ib ^ 1], gA + (size_t)(kb + 32) * 2, 32);
      tdm_load_tile(offB[ib ^ 1], gB + (size_t)(kb + 32) * 2, 256);
    }
#else
    ib = 0;                               // single-buffer cooperative fallback
    if (tid < 128) {
      int r = tid >> 2, c = tid & 3;
      *(v8bf*)&sA[0][r * 32 + c * 8] =
          *(const v8bf*)(A + (size_t)(mBase + r) * KP + kb + c * 8);
    }
#pragma unroll
    for (int j = 0; j < 4; ++j) {
      int idx = tid + j * 256;
      int r = idx >> 2, c = idx & 3;
      *(v8bf*)&sB[0][r * 32 + c * 8] =
          *(const v8bf*)(B + (size_t)(nBase + r) * KP + kb + c * 8);
    }
    __syncthreads();
#endif
    const __bf16* lA = &sA[ib][wm * 16 * 32];
    const __bf16* lB = &sB[ib][0];
    v16bf a  = load_frag16x32(lA, 32, row, hf, 0);
    v16bf b0 = load_frag16x32(lB + (wn * 64 +  0) * 32, 32, row, hf, 0);
    v16bf b1 = load_frag16x32(lB + (wn * 64 + 16) * 32, 32, row, hf, 0);
    v16bf b2 = load_frag16x32(lB + (wn * 64 + 32) * 32, 32, row, hf, 0);
    v16bf b3 = load_frag16x32(lB + (wn * 64 + 48) * 32, 32, row, hf, 0);
    acc0 = wmma_bf16(a, b0, acc0);
    acc1 = wmma_bf16(a, b1, acc1);
    acc2 = wmma_bf16(a, b2, acc2);
    acc3 = wmma_bf16(a, b3, acc3);
#if USE_TDM
    if (wave == 0) __builtin_amdgcn_s_wait_tensorcnt(0);  // next buffer landed
    __syncthreads();   // all waves done reading current buffer
#else
    __syncthreads();
#endif
  }
  int col = row;   // C/D layout: lane%16 = N, VGPR r -> M = r + 8*half
  int n0 = nBase + wn * 64;
#pragma unroll
  for (int r = 0; r < 8; r++) {
    int m = mBase + wm * 16 + r + hf * 8;
    size_t o = (size_t)m * G3 + n0 + col;
    D[o +  0] = acc0[r] + bias[n0 + col +  0];
    D[o + 16] = acc1[r] + bias[n0 + col + 16];
    D[o + 32] = acc2[r] + bias[n0 + col + 32];
    D[o + 48] = acc3[r] + bias[n0 + col + 48];
  }
}

// ---------------------------------------------------------------------------
// Main GRU: persistent single workgroup (32 waves). Batch padded 4 -> 16 rows.
// Per step: hw = hpad @ Whh^T via 96 WMMA N-tiles (3/wave), then gate VALU.
// Emits H (f32) and Hbf (bf16) for the downstream topic GEMMs.
// ---------------------------------------------------------------------------
__global__ void __launch_bounds__(1024, 1)
gru_main(const float* __restrict__ xw, const __bf16* __restrict__ whh,
         const float* __restrict__ bhh, const float* __restrict__ h0,
         float* __restrict__ H, __bf16* __restrict__ Hbf) {
  __shared__ float  sh[BS][NH];
  __shared__ __bf16 hpad[16][NH];
  __shared__ float  shw[BS][G3];
  int tid = threadIdx.x;
  for (int i = tid; i < 16 * NH; i += blockDim.x) {
    int r = i / NH, j = i - r * NH;
    float v = (r < BS) ? h0[j] : 0.0f;
    hpad[r][j] = (__bf16)v;
    if (r < BS) sh[r][j] = v;
  }
  __syncthreads();
  int wave = tid >> 5, lane = tid & 31;
  int row = lane & 15, hf = lane >> 4;
  for (int t = 0; t < NWORD; ++t) {
    for (int nt = wave * 3; nt < wave * 3 + 3; ++nt) {
      v8f acc = {};
      const __bf16* bb = whh + (size_t)nt * 16 * NH;
      for (int kt = 0; kt < NH; kt += 32) {
        v16bf a = load_frag16x32(&hpad[0][0], NH, row, hf, kt);
        v16bf b = load_frag16x32(bb, NH, row, hf, kt);
        acc = wmma_bf16(a, b, acc);
      }
      if (hf == 0) {           // only batch rows 0..3 are live
#pragma unroll
        for (int r4 = 0; r4 < 4; ++r4)
          shw[r4][nt * 16 + row] = acc[r4] + bhh[nt * 16 + row];
      }
    }
    __syncthreads();
    for (int e = tid; e < BS * NH; e += blockDim.x) {
      int b = e >> 9, j = e & (NH - 1);
      size_t xrow = ((size_t)b * NWORD + t) * G3;
      float xr = xw[xrow + j], xz = xw[xrow + NH + j], xn = xw[xrow + 2 * NH + j];
      float r = sigmoidf_(xr + shw[b][j]);
      float z = sigmoidf_(xz + shw[b][NH + j]);
      float n = tanhf(xn + r * shw[b][2 * NH + j]);
      float hnew = (1.0f - z) * n + z * sh[b][j];
      sh[b][j] = hnew;
      hpad[b][j] = (__bf16)hnew;
      size_t o = ((size_t)b * NWORD + t) * NH + j;
      H[o] = hnew;
      Hbf[o] = (__bf16)hnew;
    }
    __syncthreads();
  }
}

// ---------------------------------------------------------------------------
// Um_cat[b,k,h] (bf16, k padded 40->48): k<NT from U·mu, NT<=k<2NT from V·mv.
// mstride = 0 (broadcast m0) or NH (per-batch m1).
// ---------------------------------------------------------------------------
__global__ void um_kernel(const float* __restrict__ U, const float* __restrict__ V,
                          const float* __restrict__ mu, const float* __restrict__ mv,
                          int mstride, __bf16* __restrict__ out) {
  int idx = blockIdx.x * blockDim.x + threadIdx.x;
  if (idx >= BS * 48 * NH) return;
  int hcol = idx & (NH - 1);
  int k = (idx / NH) % 48;
  int b = idx / (48 * NH);
  float s = 0.0f;
  if (k < NT) {
    const float* u = U + ((size_t)k * NH + hcol) * NH;
    const float* m = mu + b * mstride;
    for (int j = 0; j < NH; j++) s += u[j] * m[j];
  } else if (k < 2 * NT) {
    const float* u = V + ((size_t)(k - NT) * NH + hcol) * NH;
    const float* m = mv + b * mstride;
    for (int j = 0; j < NH; j++) s += u[j] * m[j];
  }
  out[idx] = (__bf16)s;
}

// ---------------------------------------------------------------------------
// Topic product GEMM: a[b,w,k] = tanh(sum_h Hbf[b,w,h] * Um[b,k,h]).
// One wave per (mt, nt, b); M=2048/b, N=48 (store k<NV), K=512.
// ---------------------------------------------------------------------------
__global__ void tp_gemm(const __bf16* __restrict__ Hbf, const __bf16* __restrict__ Um,
                        float* __restrict__ out) {
  int mt = blockIdx.x, nt = blockIdx.y, b = blockIdx.z;
  int lane = threadIdx.x & 31;
  int row = lane & 15, hf = lane >> 4;
  const __bf16* A = Hbf + ((size_t)b * NWORD + mt * 16) * NH;
  const __bf16* B = Um + ((size_t)b * 48 + nt * 16) * NH;
  v8f acc = {};
  for (int kt = 0; kt < NH; kt += 32) {
    if (kt + 32 < NH)
      __builtin_prefetch(A + (size_t)row * NH + kt + 32, 0, 1);
    v16bf a  = load_frag16x32(A, NH, row, hf, kt);
    v16bf bb = load_frag16x32(B, NH, row, hf, kt);
    acc = wmma_bf16(a, bb, acc);
  }
  int col = nt * 16 + row;
  if (col < NV) {
#pragma unroll
    for (int r = 0; r < 8; r++) {
      int m = mt * 16 + r + hf * 8;
      out[((size_t)b * NWORD + m) * NV + col] = tanhf(acc[r]);
    }
  }
}

// xw_att = a @ Wih^T + bih   (rows=ROWS, N=GV3, K=NV)
__global__ void xw_att(const float* __restrict__ a, const float* __restrict__ Wih,
                       const float* __restrict__ bih, float* __restrict__ out) {
  int idx = blockIdx.x * blockDim.x + threadIdx.x;
  if (idx >= ROWS * GV3) return;
  int g = idx % GV3, rrow = idx / GV3;
  const float* av = a + (size_t)rrow * NV;
  const float* w  = Wih + (size_t)g * NV;
  float s = bih[g];
#pragma unroll
  for (int j = 0; j < NV; j++) s += av[j] * w[j];
  out[idx] = s;
}

// Tiny attention GRU (hidden NV=40): single workgroup sequential scan.
__global__ void att_gru(const float* __restrict__ xw, const float* __restrict__ Whh,
                        const float* __restrict__ bhh, const float* __restrict__ r0,
                        float* __restrict__ out) {
  __shared__ float st[BS][NV];
  __shared__ float shw[BS][GV3];
  int tid = threadIdx.x;
  if (tid < BS * NV) st[tid / NV][tid % NV] = r0[tid % NV];
  __syncthreads();
  for (int t = 0; t < NWORD; ++t) {
    for (int e = tid; e < BS * GV3; e += blockDim.x) {
      int b = e / GV3, g = e - b * GV3;
      float s = bhh[g];
      const float* w = Whh + g * NV;
#pragma unroll
      for (int j = 0; j < NV; j++) s += st[b][j] * w[j];
      shw[b][g] = s;
    }
    __syncthreads();
    if (tid < BS * NV) {
      int b = tid / NV, j = tid - b * NV;
      size_t xrow = ((size_t)b * NWORD + t) * GV3;
      float r = sigmoidf_(xw[xrow + j] + shw[b][j]);
      float z = sigmoidf_(xw[xrow + NV + j] + shw[b][NV + j]);
      float n = tanhf(xw[xrow + 2 * NV + j] + r * shw[b][2 * NV + j]);
      float hnew = (1.0f - z) * n + z * st[b][j];
      st[b][j] = hnew;
      out[((size_t)b * NWORD + t) * NV + j] = hnew;
    }
    __syncthreads();
  }
}

// Softmax attention over w + context vector: ctx[b,:] = sum_w softmax(ha·v)[w]*H[b,w,:]
__global__ void attn_ctx(const float* __restrict__ ha, const float* __restrict__ v,
                         const float* __restrict__ H, float* __restrict__ ctx) {
  int b = blockIdx.x, tid = threadIdx.x;
  __shared__ float s[NWORD];
  __shared__ float red[256];
  float lmax = -1e30f;
  for (int w = tid; w < NWORD; w += 256) {
    const float* hv = ha + ((size_t)b * NWORD + w) * NV;
    float d = 0.0f;
#pragma unroll
    for (int j = 0; j < NV; j++) d += hv[j] * v[j];
    s[w] = d;
    lmax = fmaxf(lmax, d);
  }
  red[tid] = lmax; __syncthreads();
  for (int o = 128; o > 0; o >>= 1) { if (tid < o) red[tid] = fmaxf(red[tid], red[tid + o]); __syncthreads(); }
  float mx = red[0]; __syncthreads();
  float lsum = 0.0f;
  for (int w = tid; w < NWORD; w += 256) { float e = __expf(s[w] - mx); s[w] = e; lsum += e; }
  red[tid] = lsum; __syncthreads();
  for (int o = 128; o > 0; o >>= 1) { if (tid < o) red[tid] += red[tid + o]; __syncthreads(); }
  float inv = 1.0f / red[0]; __syncthreads();
  for (int hc = tid; hc < NH; hc += 256) {
    float acc = 0.0f;
    for (int w = 0; w < NWORD; ++w) acc += s[w] * H[((size_t)b * NWORD + w) * NH + hc];
    ctx[b * NH + hc] = acc * inv;
  }
}

// m1[b,h] = tanh((m0 @ M)[h]) + ctx[b,h]
__global__ void mem_update(const float* __restrict__ m0, const float* __restrict__ M,
                           const float* __restrict__ ctx, float* __restrict__ m1) {
  int idx = blockIdx.x * blockDim.x + threadIdx.x;
  if (idx >= BS * NH) return;
  int h = idx & (NH - 1);
  float s = 0.0f;
  for (int j = 0; j < NH; j++) s += m0[j] * M[(size_t)j * NH + h];
  m1[idx] = tanhf(s) + ctx[idx];
}

// y = softmax((h0s+h1s) @ W^T + b) over NC
__global__ void final_softmax(const float* __restrict__ h0s, const float* __restrict__ h1s,
                              const float* __restrict__ W, const float* __restrict__ bias,
                              float* __restrict__ y) {
  int rrow = blockIdx.x * blockDim.x + threadIdx.x;
  if (rrow >= ROWS) return;
  float hs[NV];
  const float* p0 = h0s + (size_t)rrow * NV;
  const float* p1 = h1s + (size_t)rrow * NV;
#pragma unroll
  for (int j = 0; j < NV; j++) hs[j] = p0[j] + p1[j];
  float lg[NC], mx = -1e30f;
#pragma unroll
  for (int c = 0; c < NC; c++) {
    float s = bias[c];
    const float* w = W + c * NV;
    for (int j = 0; j < NV; j++) s += hs[j] * w[j];
    lg[c] = s; mx = fmaxf(mx, s);
  }
  float sum = 0.0f;
  for (int c = 0; c < NC; c++) { lg[c] = __expf(lg[c] - mx); sum += lg[c]; }
  float inv = 1.0f / sum;
  for (int c = 0; c < NC; c++) y[(size_t)rrow * NC + c] = lg[c] * inv;
}

// ---------------------------------------------------------------------------
extern "C" void kernel_launch(void* const* d_in, const int* in_sizes, int n_in,
                              void* d_out, int out_size, void* d_ws, size_t ws_size,
                              hipStream_t stream) {
  const int*   ctxw = (const int*)d_in[0];
  const int*   iemb = (const int*)d_in[1];
  const float* emb  = (const float*)d_in[4];
  const float* pad  = (const float*)d_in[5];
  const float* pkt  = (const float*)d_in[6];
  const float* h0   = (const float*)d_in[7];
  const float* m0a  = (const float*)d_in[8];
  const float* m0o  = (const float*)d_in[9];
  const float* Ua   = (const float*)d_in[10];
  const float* Va   = (const float*)d_in[11];
  const float* Uo   = (const float*)d_in[12];
  const float* Vo   = (const float*)d_in[13];
  const float* Ma   = (const float*)d_in[14];
  const float* Mo   = (const float*)d_in[15];
  const float* va   = (const float*)d_in[16];
  const float* vo   = (const float*)d_in[17];
  const float* r0a  = (const float*)d_in[18];
  const float* r0o  = (const float*)d_in[19];
  const float* gWih = (const float*)d_in[20];
  const float* gWhh = (const float*)d_in[21];
  const float* gbih = (const float*)d_in[22];
  const float* gbhh = (const float*)d_in[23];
  const float* gaWih = (const float*)d_in[24];
  const float* gaWhh = (const float*)d_in[25];
  const float* gabih = (const float*)d_in[26];
  const float* gabhh = (const float*)d_in[27];
  const float* goWih = (const float*)d_in[28];
  const float* goWhh = (const float*)d_in[29];
  const float* gobih = (const float*)d_in[30];
  const float* gobhh = (const float*)d_in[31];
  const float* laW = (const float*)d_in[32];
  const float* lab = (const float*)d_in[33];
  const float* loW = (const float*)d_in[34];
  const float* lob = (const float*)d_in[35];
  float* out = (float*)d_out;

  char* ws = (char*)d_ws;
  size_t off = 0;
  auto alloc = [&](size_t bytes) -> char* {
    char* p = ws + off; off += (bytes + 255) & ~(size_t)255; return p;
  };
  __bf16* xbf   = (__bf16*)alloc((size_t)ROWS * KP * 2);
  __bf16* wihb  = (__bf16*)alloc((size_t)G3 * KP * 2);
  __bf16* whhb  = (__bf16*)alloc((size_t)G3 * NH * 2);
  float*  xw    = (float*)alloc((size_t)ROWS * G3 * 4);
  float*  H     = (float*)alloc((size_t)ROWS * NH * 4);
  __bf16* Hbf   = (__bf16*)alloc((size_t)ROWS * NH * 2);
  __bf16* umA   = (__bf16*)alloc((size_t)BS * 48 * NH * 2);
  __bf16* umO   = (__bf16*)alloc((size_t)BS * 48 * NH * 2);
  float*  aA    = (float*)alloc((size_t)ROWS * NV * 4);
  float*  aO    = (float*)alloc((size_t)ROWS * NV * 4);
  float*  xwA   = (float*)alloc((size_t)ROWS * GV3 * 4);
  float*  xwO   = (float*)alloc((size_t)ROWS * GV3 * 4);
  float*  ha0   = (float*)alloc((size_t)ROWS * NV * 4);
  float*  ho0   = (float*)alloc((size_t)ROWS * NV * 4);
  float*  ha1   = (float*)alloc((size_t)ROWS * NV * 4);
  float*  ho1   = (float*)alloc((size_t)ROWS * NV * 4);
  float*  ca    = (float*)alloc((size_t)BS * NH * 4);
  float*  co    = (float*)alloc((size_t)BS * NH * 4);
  float*  ma1   = (float*)alloc((size_t)BS * NH * 4);
  float*  mo1   = (float*)alloc((size_t)BS * NH * 4);

  // 1) precision prep + gather
  cvt_pad_bf16<<<(G3 * KP + 255) / 256, 256, 0, stream>>>(gWih, wihb, G3, NIN, KP);
  cvt_pad_bf16<<<(G3 * NH + 255) / 256, 256, 0, stream>>>(gWhh, whhb, G3, NH, NH);
  build_x<<<ROWS, 256, 0, stream>>>(ctxw, iemb, emb, pad, pkt, xbf);

  // 2) big WMMA GEMM with TDM-staged LDS tiles: xw = x @ Wih^T + bih
  gemm_xw<<<dim3(ROWS / 32, G3 / 256), 256, 0, stream>>>(xbf, wihb, gbih, xw);

  // 3) sequential main GRU (persistent workgroup, WMMA recurrence)
  gru_main<<<1, 1024, 0, stream>>>(xw, whhb, gbhh, h0, H, Hbf);

  const int umGrid = (BS * 48 * NH + 255) / 256;
  const dim3 tpGrid(NWORD / 16, 3, BS);
  const int xaGrid = (ROWS * GV3 + 255) / 256;

  // 4) round 0: memory = m0 (broadcast)
  um_kernel<<<umGrid, 256, 0, stream>>>(Ua, Va, m0a, m0o, 0, umA);
  um_kernel<<<umGrid, 256, 0, stream>>>(Uo, Vo, m0o, m0a, 0, umO);
  tp_gemm<<<tpGrid, 32, 0, stream>>>(Hbf, umA, aA);
  tp_gemm<<<tpGrid, 32, 0, stream>>>(Hbf, umO, aO);
  xw_att<<<xaGrid, 256, 0, stream>>>(aA, gaWih, gabih, xwA);
  xw_att<<<xaGrid, 256, 0, stream>>>(aO, goWih, gobih, xwO);
  att_gru<<<1, 512, 0, stream>>>(xwA, gaWhh, gabhh, r0a, ha0);
  att_gru<<<1, 512, 0, stream>>>(xwO, goWhh, gobhh, r0o, ho0);

  // 5) attention -> context -> memory update
  attn_ctx<<<BS, 256, 0, stream>>>(ha0, va, H, ca);
  attn_ctx<<<BS, 256, 0, stream>>>(ho0, vo, H, co);
  mem_update<<<(BS * NH + 255) / 256, 256, 0, stream>>>(m0a, Ma, ca, ma1);
  mem_update<<<(BS * NH + 255) / 256, 256, 0, stream>>>(m0o, Mo, co, mo1);

  // 6) round 1: memory = m1 (per-batch)
  um_kernel<<<umGrid, 256, 0, stream>>>(Ua, Va, ma1, mo1, NH, umA);
  um_kernel<<<umGrid, 256, 0, stream>>>(Uo, Vo, mo1, ma1, NH, umO);
  tp_gemm<<<tpGrid, 32, 0, stream>>>(Hbf, umA, aA);
  tp_gemm<<<tpGrid, 32, 0, stream>>>(Hbf, umO, aO);
  xw_att<<<xaGrid, 256, 0, stream>>>(aA, gaWih, gabih, xwA);
  xw_att<<<xaGrid, 256, 0, stream>>>(aO, goWih, gobih, xwO);
  att_gru<<<1, 512, 0, stream>>>(xwA, gaWhh, gabhh, r0a, ha1);
  att_gru<<<1, 512, 0, stream>>>(xwO, goWhh, gobhh, r0o, ho1);

  // 7) output heads: ya then yo, concatenated
  final_softmax<<<(ROWS + 255) / 256, 256, 0, stream>>>(ha0, ha1, laW, lab, out);
  final_softmax<<<(ROWS + 255) / 256, 256, 0, stream>>>(ho0, ho1, loW, lob,
                                                        out + (size_t)ROWS * NC);
}
